// CausalSelfAttention_65609920414214
// MI455X (gfx1250) — matrix operand order repaired and checked by reading, not compiled, not run on hardware
//
#include <hip/hip_runtime.h>
#include <cmath>

#define DIMW 2048
#define NH 16
#define NKV 4
#define HD 128
#define SEQ 2048
#define BSZ 4
#define ROWS (BSZ * SEQ)     // 8192
#define KVDIM (NKV * HD)     // 512

typedef __bf16 bf16;
typedef __attribute__((ext_vector_type(8)))  bf16  v8bf;
typedef __attribute__((ext_vector_type(16))) bf16  v16bf;
typedef __attribute__((ext_vector_type(8)))  float v8f;

union BFrag { v16bf v; v8bf h[2]; };

__device__ __forceinline__ v8bf ld8(const bf16* p) { return *(const v8bf*)p; }

#define WMMA_BF16(a, b, c) \
    __builtin_amdgcn_wmma_f32_16x16x32_bf16(false, (a), false, (b), (short)0, (c), false, false)

// Async 16-byte global -> LDS copy (ASYNCcnt path, CDNA5 ISA).
__device__ __forceinline__ void async_copy16(uint32_t lds_off, const void* gptr) {
    asm volatile("global_load_async_to_lds_b128 %0, %1, off"
                 :: "v"(lds_off), "v"((unsigned long long)(uintptr_t)gptr)
                 : "memory");
}
#define WAIT_ASYNC(n) asm volatile("s_wait_asynccnt " #n ::: "memory")

// ---------------------------------------------------------------- converts
__global__ void k_f32_to_bf16(const float* __restrict__ src, bf16* __restrict__ dst, int n) {
    int i = blockIdx.x * blockDim.x + threadIdx.x;
    if (i < n) dst[i] = (bf16)src[i];
}

// V (row=b*S+s, col=kv*HD+d) fp32 -> vt[b][kv][d][s] bf16 (key-contiguous for B frags)
__global__ void k_convert_v(const float* __restrict__ Vf, bf16* __restrict__ vt) {
    int idx = blockIdx.x * blockDim.x + threadIdx.x;
    if (idx >= ROWS * KVDIM) return;
    int row = idx / KVDIM, col = idx % KVDIM;
    int b = row / SEQ, s = row % SEQ;
    int kv = col / HD, d = col % HD;
    vt[((size_t)(b * NKV + kv) * HD + d) * SEQ + s] = (bf16)Vf[idx];
}

// ---------------------------------------------------------------- TN GEMM (async-LDS, double buffered)
// C[M,N] = A[M,K] * B[N,K]^T ; A,B bf16 row-major, C fp32.
// Block tile 256x64 (8 waves, 32 rows each), K staged through LDS in 32-slabs.
// Per wave/slab: 2 A frags + 4 B frags (12 ds_b128) -> 8 WMMAs (B reused 2x).
#define BM 256
#define BN 64
#define BK 32
#define LDP 48   // LDS row pitch in elements (96B: 16B-aligned frags, spread banks)

__global__ __launch_bounds__(256) void k_gemm_tn(const bf16* __restrict__ A,
                                                 const bf16* __restrict__ B,
                                                 float* __restrict__ C,
                                                 int M, int N, int K) {
    __shared__ bf16 As[2][BM * LDP];   // 24 KB per buffer
    __shared__ bf16 Bs[2][BN * LDP];   //  6 KB per buffer

    const int tid  = threadIdx.x;
    const int lane = tid & 31;
    const int w    = tid >> 5;
    const int tilesN = N / BN;
    const int bm = blockIdx.x / tilesN;
    const int bn = blockIdx.x % tilesN;
    if (bm * BM >= M) return;
    const int h = lane >> 4;
    const int r = lane & 15;

    const bf16* Ablk = A + (size_t)bm * BM * K;
    const bf16* Bblk = B + (size_t)bn * BN * K;

    // copy assignments: 16B chunks (8 bf16). A: 1024 chunks, B: 256 chunks, 256 threads.
    const int ar = tid >> 2;                  // rows 0..63 (+64,+128,+192)
    const int ac = (tid & 3) * 8;             // col offset in slab
    const int br = tid >> 2;
    const int bc = ac;

    auto issue = [&](int buf, int k0) {
#pragma unroll
        for (int i = 0; i < 4; ++i)
            async_copy16((uint32_t)(uintptr_t)&As[buf][(ar + i * 64) * LDP + ac],
                         Ablk + (size_t)(ar + i * 64) * K + k0 + ac);
        async_copy16((uint32_t)(uintptr_t)&Bs[buf][br * LDP + bc],
                     Bblk + (size_t)br * K + k0 + bc);
    };

    v8f acc[2][4] = {};
    const bf16* arow0 = &As[0][(w * 32 + r) * LDP];
    const bf16* brow0 = &Bs[0][r * LDP + h * 8];

    auto compute = [&](int buf) {
        const bf16* arow = arow0 + buf * (BM * LDP);
        const bf16* brow = brow0 + buf * (BN * LDP);
        BFrag a[2], b[4];
#pragma unroll
        for (int i = 0; i < 2; ++i) {        // A rows w*32 + i*16 + {r, r}
            a[i].h[0] = ld8(arow + i * 16 * LDP + h * 8);
            a[i].h[1] = ld8(arow + i * 16 * LDP + 16 + h * 8);
        }
#pragma unroll
        for (int j = 0; j < 4; ++j) {        // batch all B-fragment loads
            b[j].h[0] = ld8(brow + j * 16 * LDP);
            b[j].h[1] = ld8(brow + j * 16 * LDP + 16);
        }
#pragma unroll
        for (int j = 0; j < 4; ++j) {        // 8 WMMAs, each B frag reused twice
            acc[0][j] = WMMA_BF16(a[0].v, b[j].v, acc[0][j]);
            acc[1][j] = WMMA_BF16(a[1].v, b[j].v, acc[1][j]);
        }
    };

    const int nst = K / BK;
    issue(0, 0);
    int i = 0;
    for (; i < nst - 1; ++i) {               // steady state: copy(i+1) overlaps math(i)
        issue((i + 1) & 1, (i + 1) * BK);
        WAIT_ASYNC(0x5);                     // our 5 stage-i ops done (in-order)
        __syncthreads();
        compute(i & 1);
        __syncthreads();                     // readers done before buffer reuse
    }
    WAIT_ASYNC(0x0);                         // peeled epilogue: branchless steady loop
    __syncthreads();
    compute(i & 1);

#pragma unroll
    for (int ii = 0; ii < 2; ++ii) {
        float* crow = C + (size_t)(bm * BM + w * 32 + ii * 16 + h * 8) * N + bn * BN;
#pragma unroll
        for (int j = 0; j < 4; ++j)
#pragma unroll
            for (int rr = 0; rr < 8; ++rr)
                crow[(size_t)rr * N + j * 16 + r] = acc[ii][j][rr];
    }
}

// ---------------------------------------------------------------- RMSNorm + RoPE (+gain)
__global__ __launch_bounds__(256) void k_norm_rope(const float* __restrict__ in,
                                                   const float* __restrict__ gain,
                                                   bf16* __restrict__ out,
                                                   int nheads, float scale) {
    const int stride = nheads * HD;
    const int lane = threadIdx.x & 31;
    const int wave = blockIdx.x * (blockDim.x >> 5) + (threadIdx.x >> 5);
    const int head = wave % nheads;
    const int row  = wave / nheads;           // b*SEQ + s
    if (row >= ROWS) return;
    const int b = row / SEQ;
    const int s = row % SEQ;

    const float4 t = *(const float4*)(in + (size_t)row * stride + head * HD + lane * 4);
    float v[4] = {t.x, t.y, t.z, t.w};
    float ss = v[0] * v[0] + v[1] * v[1] + v[2] * v[2] + v[3] * v[3];
#pragma unroll
    for (int m = 1; m < 32; m <<= 1) ss += __shfl_xor(ss, m, 32);
    const float rn = rsqrtf(ss * (1.0f / HD) + 1.1920928955078125e-07f);
#pragma unroll
    for (int j = 0; j < 4; ++j) v[j] *= rn;

    float g = scale;
    if (gain) g *= gain[head];

    bf16* dst = out + ((size_t)(b * nheads + head) * SEQ + s) * HD + lane * 4;
#pragma unroll
    for (int j = 0; j < 4; ++j) {
        const float part = __shfl_xor(v[j], 16, 32);   // element at d^64
        const int d = lane * 4 + j;
        const int f = (d < 64) ? d : (d - 64);
        const float ang = (float)s * __powf(10000.0f, -(float)f * (1.0f / 64.0f));
        float sn, c;
        __sincosf(ang, &sn, &c);
        const float o = (d < 64) ? (v[j] * c + part * sn)
                                 : (part * -sn + v[j] * c);
        dst[j] = (bf16)(o * g);
    }
}

// ---------------------------------------------------------------- flash attention (async K/V staging)
// Block = 8 waves = 8 consecutive 16-row q tiles of ONE (b, hq): K/V tiles shared via LDS.
#define KP 136   // K tile pitch (elems): 272B rows, 16B aligned, bank-spread
#define VP 40    // V tile pitch (elems): 80B rows, 16B aligned, bank-spread

__global__ __launch_bounds__(256) void k_attention(const bf16* __restrict__ Q,
                                                   const bf16* __restrict__ Kc,
                                                   const bf16* __restrict__ Vt,
                                                   bf16* __restrict__ Y) {
    __shared__ bf16 Ks[2][32 * KP];     // 32 keys x 128 dims (padded)
    __shared__ bf16 Vs[2][128 * VP];    // 128 dims x 32 keys (padded)
    __shared__ bf16 ptile[8][16 * 32];  // per-wave P tile

    const int tid  = threadIdx.x;
    const int lane = tid & 31;
    const int wv   = tid >> 5;
    const int qtb  = blockIdx.x % (SEQ / 128);       // 128-row q block
    const int bh   = blockIdx.x / (SEQ / 128);
    const int hq   = bh % NH;
    const int b    = bh / NH;
    if (b >= BSZ) return;
    const int kv = hq / (NH / NKV);
    const int qt = qtb * 8 + wv;                     // this wave's 16-row q tile
    const int h = lane >> 4;
    const int r = lane & 15;

    const bf16* kbase = Kc + (size_t)(b * NKV + kv) * SEQ * HD;
    const bf16* vbase = Vt + (size_t)(b * NKV + kv) * HD * SEQ;

    // Q A-fragments: 16 rows x 128 K, 4 chunks of 32 (loaded once from global)
    const bf16* qbase = Q + ((size_t)(b * NH + hq) * SEQ + qt * 16 + r) * HD;
    BFrag qa[4];
#pragma unroll
    for (int f = 0; f < 4; ++f) {
        qa[f].h[0] = ld8(qbase + f * 32 + h * 8);
        qa[f].h[1] = ld8(qbase + f * 32 + 16 + h * 8);
    }

    // async copy assignments: K tile 512 chunks, V tile 512 chunks; 2+2 per thread
    const int kr0 = tid >> 3, kc = (tid & 7) * 8;
    const int vr0 = tid >> 2, vc = (tid & 3) * 8;
    auto issue = [&](int buf, int base) {
        async_copy16((uint32_t)(uintptr_t)&Ks[buf][kr0 * KP + kc],
                     kbase + (size_t)(base + kr0) * HD + kc);
        async_copy16((uint32_t)(uintptr_t)&Ks[buf][(kr0 + 16) * KP + kc],
                     kbase + (size_t)(base + kr0 + 16) * HD + kc);
        async_copy16((uint32_t)(uintptr_t)&Vs[buf][vr0 * VP + vc],
                     vbase + (size_t)vr0 * SEQ + base + vc);
        async_copy16((uint32_t)(uintptr_t)&Vs[buf][(vr0 + 64) * VP + vc],
                     vbase + (size_t)(vr0 + 64) * SEQ + base + vc);
    };

    v8f o[8] = {};
    float mrow[8], lrow[8];
#pragma unroll
    for (int rr = 0; rr < 8; ++rr) { mrow[rr] = -INFINITY; lrow[rr] = 0.0f; }

    auto compute = [&](int buf, int base) {
        if (qt * 16 + 15 < base) return;             // wave-uniform causal skip (EXEC intact)
        v8f s0 = {}, s1 = {};
        BFrag kf[8];
#pragma unroll
        for (int f = 0; f < 4; ++f) {                // batch all K-fragment LDS loads
            kf[2 * f + 0].h[0] = ld8(&Ks[buf][r * KP + f * 32 + h * 8]);
            kf[2 * f + 0].h[1] = ld8(&Ks[buf][r * KP + f * 32 + 16 + h * 8]);
            kf[2 * f + 1].h[0] = ld8(&Ks[buf][(16 + r) * KP + f * 32 + h * 8]);
            kf[2 * f + 1].h[1] = ld8(&Ks[buf][(16 + r) * KP + f * 32 + 16 + h * 8]);
        }
#pragma unroll
        for (int f = 0; f < 4; ++f) {                // S = Q K^T over dim 128
            s0 = WMMA_BF16(qa[f].v, kf[2 * f + 0].v, s0);
            s1 = WMMA_BF16(qa[f].v, kf[2 * f + 1].v, s1);
        }
        const int key0 = base + r;
        const int key1 = base + 16 + r;
#pragma unroll
        for (int rr = 0; rr < 8; ++rr) {             // causal mask + online softmax
            const int qg = qt * 16 + rr + h * 8;
            float a0 = (key0 <= qg) ? s0[rr] : -INFINITY;
            float a1 = (key1 <= qg) ? s1[rr] : -INFINITY;
            float tm = fmaxf(a0, a1);
#pragma unroll
            for (int m = 1; m < 16; m <<= 1) tm = fmaxf(tm, __shfl_xor(tm, m, 32));
            const float nm = fmaxf(mrow[rr], tm);
            const float corr = __expf(mrow[rr] - nm);
            const float p0 = __expf(a0 - nm);
            const float p1 = __expf(a1 - nm);
            float ps = p0 + p1;
#pragma unroll
            for (int m = 1; m < 16; m <<= 1) ps += __shfl_xor(ps, m, 32);
            lrow[rr] = lrow[rr] * corr + ps;
            mrow[rr] = nm;
#pragma unroll
            for (int j = 0; j < 8; ++j) o[j][rr] *= corr;
            ptile[wv][(rr + h * 8) * 32 + r]      = (bf16)p0;
            ptile[wv][(rr + h * 8) * 32 + 16 + r] = (bf16)p1;
        }
        asm volatile("s_wait_dscnt 0" ::: "memory"); // P visible across lanes (wave-sync)
        BFrag pa, vb[8];
        pa.h[0] = *(const v8bf*)&ptile[wv][r * 32 + h * 8];
        pa.h[1] = *(const v8bf*)&ptile[wv][r * 32 + 16 + h * 8];
#pragma unroll
        for (int j = 0; j < 8; ++j) {                // batch all V-fragment LDS loads
            vb[j].h[0] = ld8(&Vs[buf][(j * 16 + r) * VP + h * 8]);
            vb[j].h[1] = ld8(&Vs[buf][(j * 16 + r) * VP + 16 + h * 8]);
        }
#pragma unroll
        for (int j = 0; j < 8; ++j)                  // O += P V over 128 head dims
            o[j] = WMMA_BF16(pa.v, vb[j].v, o[j]);
    };

    const int nkt = qtb * 4 + 4;                     // 32-key tiles for the whole block
    issue(0, 0);
    int kt = 0;
    for (; kt < nkt - 1; ++kt) {
        issue((kt + 1) & 1, (kt + 1) * 32);
        WAIT_ASYNC(0x4);
        __syncthreads();
        compute(kt & 1, kt * 32);
        __syncthreads();
    }
    WAIT_ASYNC(0x0);
    __syncthreads();
    compute(kt & 1, kt * 32);

    // normalize and store bf16 at (b*S + s, hq*HD + d)
    bf16* ybase = Y + ((size_t)(b * SEQ + qt * 16 + h * 8)) * DIMW + hq * HD;
#pragma unroll
    for (int rr = 0; rr < 8; ++rr) {
        const float inv = 1.0f / lrow[rr];
#pragma unroll
        for (int j = 0; j < 8; ++j)
            ybase[(size_t)rr * DIMW + j * 16 + r] = (bf16)(o[j][rr] * inv);
    }
}

// ---------------------------------------------------------------- launch
extern "C" void kernel_launch(void* const* d_in, const int* in_sizes, int n_in,
                              void* d_out, int out_size, void* d_ws, size_t ws_size,
                              hipStream_t stream) {
    (void)in_sizes; (void)n_in; (void)out_size; (void)ws_size;
    const float* x  = (const float*)d_in[0];
    const float* Wq = (const float*)d_in[1];
    const float* Wk = (const float*)d_in[2];
    const float* Wv = (const float*)d_in[3];
    const float* Wp = (const float*)d_in[4];
    const float* qg = (const float*)d_in[5];
    float* out = (float*)d_out;

    char* ws = (char*)d_ws;
    size_t off = 0;
    auto alloc = [&](size_t bytes) -> void* {
        void* p = ws + off;
        off = (off + bytes + 255) & ~(size_t)255;
        return p;
    };

    bf16*  xb  = (bf16*)alloc((size_t)ROWS * DIMW * 2);
    bf16*  wqb = (bf16*)alloc((size_t)DIMW * DIMW * 2);
    bf16*  wkb = (bf16*)alloc((size_t)KVDIM * DIMW * 2);
    bf16*  wvb = (bf16*)alloc((size_t)KVDIM * DIMW * 2);
    bf16*  wpb = (bf16*)alloc((size_t)DIMW * DIMW * 2);
    float* Qf  = (float*)alloc((size_t)ROWS * DIMW * 4);
    float* Kf  = (float*)alloc((size_t)ROWS * KVDIM * 4);
    float* Vf  = (float*)alloc((size_t)ROWS * KVDIM * 4);
    bf16*  qb  = (bf16*)alloc((size_t)ROWS * DIMW * 2);
    bf16*  kb  = (bf16*)alloc((size_t)ROWS * KVDIM * 2);
    bf16*  vt  = (bf16*)alloc((size_t)ROWS * KVDIM * 2);
    bf16*  yb  = (bf16*)Qf;   // Qf dead after k_norm_rope(q) -> reuse for attention output

    const int T = 256;
    k_f32_to_bf16<<<(ROWS * DIMW + T - 1) / T, T, 0, stream>>>(x, xb, ROWS * DIMW);
    k_f32_to_bf16<<<(DIMW * DIMW + T - 1) / T, T, 0, stream>>>(Wq, wqb, DIMW * DIMW);
    k_f32_to_bf16<<<(KVDIM * DIMW + T - 1) / T, T, 0, stream>>>(Wk, wkb, KVDIM * DIMW);
    k_f32_to_bf16<<<(KVDIM * DIMW + T - 1) / T, T, 0, stream>>>(Wv, wvb, KVDIM * DIMW);
    k_f32_to_bf16<<<(DIMW * DIMW + T - 1) / T, T, 0, stream>>>(Wp, wpb, DIMW * DIMW);

    // QKV projections (TN GEMMs, bf16 WMMA via async-LDS pipeline, fp32 out)
    k_gemm_tn<<<(ROWS / BM) * (DIMW / BN), T, 0, stream>>>(xb, wqb, Qf, ROWS, DIMW, DIMW);
    k_gemm_tn<<<(ROWS / BM) * (KVDIM / BN), T, 0, stream>>>(xb, wkb, Kf, ROWS, KVDIM, DIMW);
    k_gemm_tn<<<(ROWS / BM) * (KVDIM / BN), T, 0, stream>>>(xb, wvb, Vf, ROWS, KVDIM, DIMW);

    // RMSNorm + RoPE; q also gets gain and the 1/sqrt(HD) softmax scale folded in
    k_norm_rope<<<ROWS * NH / 8, T, 0, stream>>>(Qf, qg, qb, NH, 0.08838834764831845f);
    k_norm_rope<<<ROWS * NKV / 8, T, 0, stream>>>(Kf, nullptr, kb, NKV, 1.0f);
    k_convert_v<<<(ROWS * KVDIM + T - 1) / T, T, 0, stream>>>(Vf, vt);

    // causal flash attention (GQA, async K/V staging, bf16 WMMA)
    k_attention<<<BSZ * NH * (SEQ / 128), T, 0, stream>>>(qb, kb, vt, yb);

    // output projection straight into d_out (fp32)
    k_gemm_tn<<<(ROWS / BM) * (DIMW / BN), T, 0, stream>>>(yb, wpb, out, ROWS, DIMW, DIMW);
}
